// Style2ResidualBlock_54966991454476
// MI455X (gfx1250) — matrix operand order repaired and checked by reading, not compiled, not run on hardware
//
#include <hip/hip_runtime.h>
#include <hip/hip_bf16.h>

// ---------------- problem constants ----------------
#define BATCH 16
#define CIN   256
#define COUT  256
#define HH    64
#define WW    64
#define PH    66                 // padded height (1px halo)
#define PW    66                 // padded width
#define KK    9                  // 3x3
#define KDIM  (CIN * KK)         // 2304
#define SDIM  128

#define LIN_SCALE  0.08838834764831845f          // 1/sqrt(128)
#define CONV_SCALE 0.020833333333333332f         // 1/sqrt(256*9)

// LDS staging: 4 padded rows x 66 px x 32 cin bf16 = 16896 B per buffer
#define BCHUNKS   1056                           // 16-byte chunks per buffer
#define BUF_BYTES (BCHUNKS * 16)                 // 16896
#define BUF_ELEMS (BUF_BYTES / 2)                // 8448 bf16
#define DUMMY_OFF (2 * BUF_BYTES)                // dummy 16B slot after both buffers

// ---------------- vector types ----------------
typedef __attribute__((ext_vector_type(16))) __bf16 v16bf;
typedef __attribute__((ext_vector_type(8)))  __bf16 v8bf;
typedef __attribute__((ext_vector_type(8)))  float  v8f;

union Frag {
    v16bf v;
    v8bf  h[2];
    unsigned int u[8];
};

__device__ __forceinline__ __bf16 f2bf(float f) {
    union { float f; unsigned u; } in;
    in.f = f;
    unsigned r = in.u + 0x7FFFu + ((in.u >> 16) & 1u);   // round-nearest-even
    unsigned short hs = (unsigned short)(r >> 16);
    union { unsigned short s; __bf16 b; } out;
    out.s = hs;
    return out.b;
}

// async global -> LDS copy of one 16-byte chunk (CDNA5, tracked by ASYNCcnt)
__device__ __forceinline__ void async_b128(unsigned lds_addr, const __bf16* gptr) {
    asm volatile("global_load_async_to_lds_b128 %0, %1, off"
                 :: "v"(lds_addr), "v"(gptr)
                 : "memory");
}

// ---------------- kernel 1: style linear  s[b,ci] ----------------
__global__ void style_lin_kernel(const float* __restrict__ c_trg,
                                 const float* __restrict__ style_w,
                                 const float* __restrict__ style_b,
                                 float* __restrict__ s) {
    int i = blockIdx.x * blockDim.x + threadIdx.x;
    if (i >= BATCH * CIN) return;
    int b = i / CIN, c = i % CIN;
    const float* ct = c_trg + b * SDIM;
    const float* sw = style_w + c * SDIM;
    float acc = 0.f;
#pragma unroll 4
    for (int j = 0; j < SDIM; ++j) acc += ct[j] * sw[j];
    s[i] = acc * LIN_SCALE + style_b[c];
}

// ---------------- kernel 2: W2[co,ci] = sum_k w^2 ; wbf[co][kk*CIN+ci] = bf16(w) ----------------
__global__ void wprep_kernel(const float* __restrict__ weight,
                             float* __restrict__ W2,
                             __bf16* __restrict__ wbf) {
    int i = blockIdx.x * blockDim.x + threadIdx.x;
    if (i >= COUT * CIN) return;
    int co = i / CIN, ci = i % CIN;
    const float* wp = weight + (size_t)i * KK;   // [0][co][ci][kh][kw]
    float sum = 0.f;
#pragma unroll
    for (int k = 0; k < KK; ++k) {
        float v = wp[k];
        sum += v * v;
        wbf[(size_t)co * KDIM + k * CIN + ci] = f2bf(v);
    }
    W2[i] = sum;
}

// ---------------- kernel 3: dscale[b,co] ----------------
__global__ void demod_kernel(const float* __restrict__ s,
                             const float* __restrict__ W2,
                             float* __restrict__ dscale) {
    int i = blockIdx.x * blockDim.x + threadIdx.x;
    if (i >= BATCH * COUT) return;
    int b = i / COUT, co = i % COUT;
    const float* sb = s + b * CIN;
    const float* w2 = W2 + (size_t)co * CIN;
    float acc = 0.f;
#pragma unroll 4
    for (int c = 0; c < CIN; ++c) {
        float sv = sb[c];
        acc += sv * sv * w2[c];
    }
    dscale[i] = CONV_SCALE * rsqrtf(CONV_SCALE * CONV_SCALE * acc + 1e-8f);
}

// ---------------- kernel 4: xs padded-NHWC bf16 = x NCHW * s  (LDS transpose) ----------------
__global__ __launch_bounds__(256) void xscale_kernel(const float* __restrict__ x,
                                                     const float* __restrict__ s,
                                                     __bf16* __restrict__ xs) {
    // grid: (CIN/32, H, B); block 256; tile = 32 cin x 64 w
    __shared__ __bf16 tile[32 * 64];
    int c0 = blockIdx.x * 32;
    int h  = blockIdx.y;
    int b  = blockIdx.z;
    int t  = threadIdx.x;
#pragma unroll
    for (int j = 0; j < 8; ++j) {
        int e  = t + 256 * j;
        int ci = e >> 6, w = e & 63;                  // coalesced along w
        float v = x[(((size_t)b * CIN + c0 + ci) * HH + h) * WW + w] * s[b * CIN + c0 + ci];
        tile[ci * 64 + w] = f2bf(v);
    }
    __syncthreads();
#pragma unroll
    for (int j = 0; j < 8; ++j) {
        int o  = t + 256 * j;
        int ci = o & 31, w = o >> 5;                  // coalesced along ci
        xs[(((size_t)b * PH + (h + 1)) * PW + (w + 1)) * CIN + c0 + ci] = tile[ci * 64 + w];
    }
}

// ---------------- kernel 5: implicit-GEMM conv, bf16 WMMA + async-LDS double buffering ----
// grid: (H/2, COUT/128, B); block 256 = 8 waves: 4 in M, 2 rows in N
// per wave: 32(M) x 64(N, one image row) = 2x4 tiles of v_wmma_f32_16x16x32_bf16
__global__ __launch_bounds__(256) void conv_wmma_kernel(
        const __bf16* __restrict__ xs,      // [B][PH][PW][CIN] bf16 (zero halo)
        const __bf16* __restrict__ wbf,     // [COUT][KK*CIN] bf16
        const float*  __restrict__ dscale,  // [B][COUT]
        float* __restrict__ out) {          // [B][COUT][H][W]
    __shared__ __align__(16) __bf16 smem[(2 * BUF_BYTES + 16) / 2];

    const int t      = threadIdx.x;
    const int lane   = t & 31;
    const int wave   = t >> 5;
    const int h0     = blockIdx.x * 2;
    const int mblk   = blockIdx.y;
    const int b      = blockIdx.z;

    const int wm     = wave & 3;          // 0..3: M position
    const int wr     = wave >> 2;         // 0..1: image row within the pair
    const int m_base = mblk * 128 + wm * 32;

    const int l15    = lane & 15;
    const int hiHalf = lane >> 4;         // 0: lanes 0-15, 1: lanes 16-31
    const int qA     = hiHalf * 8;        // A fragment K sub-offset (ISA layout)
    const int qB     = hiHalf * 8;        // B fragment K sub-offset (ISA layout)

    const unsigned ldsBase = (unsigned)(size_t)&smem[0];

    // global base of staged region: padded rows h0..h0+3, w=0, cin=0
    const __bf16* gstage = xs + (((size_t)b * PH + h0) * PW) * CIN;

    // per-lane A base pointers (immediate offsets cover kk*CIN and +16)
    const __bf16* a0p = wbf + (size_t)(m_base + l15) * KDIM + qA;
    const __bf16* a1p = wbf + (size_t)(m_base + 16 + l15) * KDIM + qA;

    v8f acc[2][4] = {};

    // ---- stage chunk c0=0 into buffer 0 ----
    {
#pragma unroll
        for (int j = 0; j < 5; ++j) {
            int q = t + 256 * j;
            int valid = q < BCHUNKS;
            int qc = valid ? q : 0;
            unsigned lds = valid ? (ldsBase + (unsigned)q * 16u) : (ldsBase + DUMMY_OFF);
            int row = qc / 264;            // 264 = PW*4 sixteen-byte chunks per row
            int rem = qc - row * 264;
            int w   = rem >> 2;
            int sub = rem & 3;
            async_b128(lds, gstage + ((size_t)row * PW + w) * CIN + sub * 8);
        }
    }

    for (int c8 = 0; c8 < 8; ++c8) {
        const int c0 = c8 * 32;
        __syncthreads();   // all waves done reading buffer (c8^1) from previous iteration
        if (c8 + 1 < 8) {
            const int cn = c0 + 32;
            const unsigned bufoff = (unsigned)((c8 + 1) & 1) * BUF_BYTES;
#pragma unroll
            for (int j = 0; j < 5; ++j) {
                int q = t + 256 * j;
                int valid = q < BCHUNKS;
                int qc = valid ? q : 0;
                unsigned lds = valid ? (ldsBase + bufoff + (unsigned)q * 16u)
                                     : (ldsBase + DUMMY_OFF);
                int row = qc / 264;
                int rem = qc - row * 264;
                int w   = rem >> 2;
                int sub = rem & 3;
                async_b128(lds, gstage + ((size_t)row * PW + w) * CIN + cn + sub * 8);
            }
            asm volatile("s_wait_asynccnt 0x5" ::: "memory");  // current buffer complete
        } else {
            asm volatile("s_wait_asynccnt 0x0" ::: "memory");
        }
        __syncthreads();   // publish staged data across waves

        const __bf16* bb = smem + (size_t)(c8 & 1) * BUF_ELEMS;
        const __bf16* a0 = a0p + c0;
        const __bf16* a1 = a1p + c0;

#pragma unroll
        for (int kk = 0; kk < KK; ++kk) {
            const int dh = kk / 3 - 1;
            const int dw = kk % 3 - 1;
            const int r  = wr + dh + 1;      // staged row index 0..3

            Frag a[2];
            a[0].h[0] = *(const v8bf*)(a0 + kk * CIN);
            a[0].h[1] = *(const v8bf*)(a0 + kk * CIN + 16);
            a[1].h[0] = *(const v8bf*)(a1 + kk * CIN);
            a[1].h[1] = *(const v8bf*)(a1 + kk * CIN + 16);

            Frag bf[4];
#pragma unroll
            for (int ni = 0; ni < 4; ++ni) {
                const int wp = ni * 16 + l15 + dw + 1;    // padded w: 0..65
                const __bf16* p = bb + ((size_t)(r * PW + wp)) * 32 + qB;
                bf[ni].h[0] = *(const v8bf*)(p);
                bf[ni].h[1] = *(const v8bf*)(p + 16);
            }

#pragma unroll
            for (int mi = 0; mi < 2; ++mi)
#pragma unroll
                for (int ni = 0; ni < 4; ++ni)
                    acc[mi][ni] = __builtin_amdgcn_wmma_f32_16x16x32_bf16(
                        false, a[mi].v, false, bf[ni].v,
                        (short)0, acc[mi][ni], false, false);
        }
    }

    // --- epilogue: scale by demod, store fp32 NCHW ---
    const int h    = h0 + wr;
    const int mrow = hiHalf * 8;
#pragma unroll
    for (int mi = 0; mi < 2; ++mi) {
#pragma unroll
        for (int rr = 0; rr < 8; ++rr) {
            const int m = m_base + mi * 16 + mrow + rr;
            const float dsc = dscale[b * COUT + m];
#pragma unroll
            for (int ni = 0; ni < 4; ++ni) {
                const int w = ni * 16 + l15;
                out[(((size_t)b * COUT + m) * HH + h) * WW + w] = acc[mi][ni][rr] * dsc;
            }
        }
    }
}

// ---------------- launch ----------------
extern "C" void kernel_launch(void* const* d_in, const int* in_sizes, int n_in,
                              void* d_out, int out_size, void* d_ws, size_t ws_size,
                              hipStream_t stream) {
    const float* x       = (const float*)d_in[0];
    const float* c_src   = (const float*)d_in[1]; (void)c_src;   // unused by reference
    const float* c_trg   = (const float*)d_in[2];
    const float* style_w = (const float*)d_in[3];
    const float* style_b = (const float*)d_in[4];
    const float* weight  = (const float*)d_in[5];
    float* out = (float*)d_out;
    (void)in_sizes; (void)n_in; (void)out_size; (void)ws_size;

    char* ws = (char*)d_ws;
    size_t off = 0;
    float* s      = (float*)(ws + off); off += (size_t)BATCH * CIN * sizeof(float);
    off = (off + 255) & ~(size_t)255;
    float* W2     = (float*)(ws + off); off += (size_t)COUT * CIN * sizeof(float);
    off = (off + 255) & ~(size_t)255;
    float* dscale = (float*)(ws + off); off += (size_t)BATCH * COUT * sizeof(float);
    off = (off + 255) & ~(size_t)255;
    __bf16* wbf   = (__bf16*)(ws + off); off += (size_t)COUT * KDIM * sizeof(__bf16);
    off = (off + 255) & ~(size_t)255;
    const size_t xs_bytes = (size_t)BATCH * PH * PW * CIN * sizeof(__bf16);  // ~35.7 MB
    __bf16* xs    = (__bf16*)(ws + off); off += xs_bytes;

    // zero the halo (and everything else) each call: capture-safe async memset
    hipMemsetAsync(xs, 0, xs_bytes, stream);

    style_lin_kernel<<<(BATCH * CIN + 255) / 256, 256, 0, stream>>>(c_trg, style_w, style_b, s);
    wprep_kernel<<<(COUT * CIN + 255) / 256, 256, 0, stream>>>(weight, W2, wbf);
    demod_kernel<<<(BATCH * COUT + 255) / 256, 256, 0, stream>>>(s, W2, dscale);
    xscale_kernel<<<dim3(CIN / 32, HH, BATCH), 256, 0, stream>>>(x, s, xs);
    conv_wmma_kernel<<<dim3(HH / 2, COUT / 128, BATCH), 256, 0, stream>>>(xs, wbf, dscale, out);
}
